// EmbeddingExtractor_25297357373570
// MI455X (gfx1250) — matrix operand order repaired and checked by reading, not compiled, not run on hardware
//
#include <hip/hip_runtime.h>
#include <stdint.h>

typedef __attribute__((ext_vector_type(16))) __bf16 v16bf;
typedef __attribute__((ext_vector_type(2)))  __bf16 v2bf;
typedef __attribute__((ext_vector_type(8)))  float  v8f;

union Frag {
  v16bf v;
  uint4 q[2];
};

// pack two f32 into two bf16 (one v_cvt_pk_bf16_f32 on CDNA5)
__device__ __forceinline__ unsigned int pkbf(float lo, float hi) {
#if __has_builtin(__builtin_amdgcn_cvt_pk_bf16_f32)
  union { v2bf v; unsigned int u; } r;
  r.v = __builtin_amdgcn_cvt_pk_bf16_f32(lo, hi);
  return r.u;
#else
  union { float f; unsigned int u; } a, b;
  a.f = lo; b.f = hi;
  return __builtin_amdgcn_perm(b.u, a.u, 0x07060302u);  // truncate-pack
#endif
}

// build a 16-element bf16 A-fragment from 16 f32 values (scaled by s)
// p points at this lane's k-run; layout: k+0..7 -> q[0], k+16..23 -> q[1]
__device__ __forceinline__ void frag_from_f32(Frag& f, const float* p, float s) {
  float4 x0 = *(const float4*)(p);
  float4 x1 = *(const float4*)(p + 4);
  float4 x2 = *(const float4*)(p + 16);
  float4 x3 = *(const float4*)(p + 20);
  f.q[0].x = pkbf(x0.x * s, x0.y * s); f.q[0].y = pkbf(x0.z * s, x0.w * s);
  f.q[0].z = pkbf(x1.x * s, x1.y * s); f.q[0].w = pkbf(x1.z * s, x1.w * s);
  f.q[1].x = pkbf(x2.x * s, x2.y * s); f.q[1].y = pkbf(x2.z * s, x2.w * s);
  f.q[1].z = pkbf(x3.x * s, x3.y * s); f.q[1].w = pkbf(x3.z * s, x3.w * s);
}

// ---------------------------------------------------------------- utilities

__global__ void zero_f32_kernel(float* __restrict__ p, long long n) {
  long long i = (long long)blockIdx.x * blockDim.x + threadIdx.x;
  if (i < n) p[i] = 0.0f;
}

// converts PAIRS of f32 -> packed bf16 (n2 = n/2 pairs)
__global__ void cvt_bf16_kernel(const float* __restrict__ w,
                                unsigned int* __restrict__ o, int n2) {
  int i = blockIdx.x * blockDim.x + threadIdx.x;
  if (i < n2) {
    float2 v = *(const float2*)(w + 2 * (size_t)i);
    o[i] = pkbf(v.x, v.y);
  }
}

__global__ void degree_kernel(const long long* __restrict__ dst,
                              float* __restrict__ deg, int E) {
  int e = blockIdx.x * blockDim.x + threadIdx.x;
  if (e < E) atomicAdd(&deg[dst[e]], 1.0f);
}

__global__ void invdeg_kernel(float* __restrict__ deg, int N) {
  int i = blockIdx.x * blockDim.x + threadIdx.x;
  if (i < N) deg[i] = 1.0f / fmaxf(deg[i], 1.0f);
}

// one thread per (edge, 4 channels); agg (25.6 MB) is L2-resident (192 MB L2)
__global__ void scatter_add_kernel(const float* __restrict__ h,
                                   const long long* __restrict__ src,
                                   const long long* __restrict__ dst,
                                   float* __restrict__ agg, int total) {
  int i = blockIdx.x * blockDim.x + threadIdx.x;
  if (i >= total) return;
  int e  = i >> 5;
  int c4 = (i & 31) << 2;
  int s = (int)src[e];
  int d = (int)dst[e];
  float4 v = *(const float4*)(h + (size_t)s * 128 + c4);
  float* a = agg + (size_t)d * 128 + c4;
  atomicAdd(a + 0, v.x);
  atomicAdd(a + 1, v.y);
  atomicAdd(a + 2, v.z);
  atomicAdd(a + 3, v.w);
}

// ---------------------------------------------------- fused SAGE layer GEMM
// out[n, :] = act( (agg[n]*invdeg[n]) @ wn^T + h[n] @ wr^T + b ),  K = 128.
// One wave (32 threads) per block; each wave owns a 16-row slab.
// Staging: raw f32 rows DMA'd into LDS via global_load_async_to_lds_b128
// (ASYNCcnt-tracked, no VGPR transit, no workgroup barrier needed).

template<int COUT, bool RELU>
__global__ __launch_bounds__(32)
void sage_wmma_kernel(const float* __restrict__ agg,
                      const float* __restrict__ hin,
                      const float* __restrict__ invdeg,
                      const unsigned short* __restrict__ wn,   // [COUT,128] bf16
                      const unsigned short* __restrict__ wr,   // [COUT,128] bf16
                      const float* __restrict__ bias,          // [COUT]
                      float* __restrict__ out, int N) {
  // [mat][row][col], +4 f32 pad -> 132-dword row stride (4-bank stagger)
  __shared__ __align__(16) float smem[2][16][132];

  const int lane = threadIdx.x & 31;
  const int half = lane >> 4;   // 0: lanes 0-15, 1: lanes 16-31
  const int l16  = lane & 15;
  const int rowBase = blockIdx.x * 16;

  const unsigned sbase  = (unsigned)(uintptr_t)&smem[0][0][0];
  const unsigned laneBy = (unsigned)(lane << 4);          // lane * 16 bytes
  const bool fullM = (rowBase + 16 <= N);

  // ---- async-stage 2 x 16 rows x 512B (one instruction moves one full row)
  if (fullM) {
#pragma unroll
    for (int t = 0; t < 32; ++t) {
      const int mat = t >> 4;
      const int row = t & 15;
      const unsigned laddr = sbase + (unsigned)((mat * 16 + row) * 132 * 4) + laneBy;
      const unsigned voff  = (unsigned)(rowBase + row) * 512u + laneBy;
      const float* bp = mat ? hin : agg;
      asm volatile("global_load_async_to_lds_b128 %0, %1, %2"
                   :: "v"(laddr), "v"(voff), "s"(bp) : "memory");
    }
  } else {
#pragma unroll
    for (int t = 0; t < 32; ++t) {
      const int mat = t >> 4;
      const int row = t & 15;
      const int rg  = rowBase + row;
      if (rg < N) {
        const unsigned laddr = sbase + (unsigned)((mat * 16 + row) * 132 * 4) + laneBy;
        const unsigned voff  = (unsigned)rg * 512u + laneBy;
        const float* bp = mat ? hin : agg;
        asm volatile("global_load_async_to_lds_b128 %0, %1, %2"
                     :: "v"(laddr), "v"(voff), "s"(bp) : "memory");
      } else {
        *(float4*)&smem[mat][row][lane << 2] = make_float4(0.f, 0.f, 0.f, 0.f);
      }
    }
  }
  asm volatile("s_wait_asynccnt 0x0" ::: "memory");

  // per-lane row scale for the aggregate operand (lane owns A-row l16)
  const int arow = rowBase + l16;
  const float inv = (arow < N) ? invdeg[arow] : 1.0f;

  // ---- A fragments (16-bit A 16x32 layout): lane half selects k+0/ k+8 runs
  Frag a[2][4];
#pragma unroll
  for (int kb = 0; kb < 4; ++kb) {
    const int c0 = kb * 32 + half * 8;
    frag_from_f32(a[0][kb], &smem[0][l16][c0], inv);
    frag_from_f32(a[1][kb], &smem[1][l16][c0], 1.0f);
  }

#pragma unroll
  for (int tn = 0; tn < COUT / 16; ++tn) {
    v8f acc = {};
    const int ocol = tn * 16 + l16;                 // this lane's output column
    const unsigned short* wnrow = wn + (size_t)ocol * 128;
    const unsigned short* wrrow = wr + (size_t)ocol * 128;
#pragma unroll
    for (int kb = 0; kb < 4; ++kb) {
      // B 32x16 layout: lane = column; lanes 0-15 hold K kb+0..7,16..23,
      // lanes 16-31 hold K kb+8..15,24..31 (two 8-bf16 runs along K).
      const int c0 = kb * 32 + half * 8;
      Frag bn, br;
      bn.q[0] = *(const uint4*)(wnrow + c0);
      bn.q[1] = *(const uint4*)(wnrow + c0 + 16);
      br.q[0] = *(const uint4*)(wrrow + c0);
      br.q[1] = *(const uint4*)(wrrow + c0 + 16);
      acc = __builtin_amdgcn_wmma_f32_16x16x32_bf16(
          false, a[0][kb].v, false, bn.v, (short)0, acc, false, false);
      acc = __builtin_amdgcn_wmma_f32_16x16x32_bf16(
          false, a[1][kb].v, false, br.v, (short)0, acc, false, false);
    }
    const float bv = bias[ocol];
    // C/D layout: VGPR v -> M = half*8 + v, N = l16
    if (fullM) {
#pragma unroll
      for (int v = 0; v < 8; ++v) {
        float val = acc[v] + bv;
        if (RELU) val = fmaxf(val, 0.0f);
        out[(size_t)(rowBase + half * 8 + v) * COUT + ocol] = val;
      }
    } else {
#pragma unroll
      for (int v = 0; v < 8; ++v) {
        const int rg = rowBase + half * 8 + v;
        if (rg < N) {
          float val = acc[v] + bv;
          if (RELU) val = fmaxf(val, 0.0f);
          out[(size_t)rg * COUT + ocol] = val;
        }
      }
    }
  }
}

// ------------------------------------------------------------------ driver

extern "C" void kernel_launch(void* const* d_in, const int* in_sizes, int n_in,
                              void* d_out, int out_size, void* d_ws, size_t ws_size,
                              hipStream_t stream) {
  const float*     x   = (const float*)d_in[0];
  const long long* ei  = (const long long*)d_in[1];
  const float*     w1n = (const float*)d_in[2];
  const float*     w1r = (const float*)d_in[3];
  const float*     b1  = (const float*)d_in[4];
  const float*     w2n = (const float*)d_in[5];
  const float*     w2r = (const float*)d_in[6];
  const float*     b2  = (const float*)d_in[7];
  const float*     w3n = (const float*)d_in[8];
  const float*     w3r = (const float*)d_in[9];
  const float*     b3  = (const float*)d_in[10];

  const int N = in_sizes[0] / 128;     // 50000
  const int E = in_sizes[1] / 2;       // 600000

  const long long* src = ei;
  const long long* dst = ei + E;

  auto al = [](size_t v) { return (v + 255) & ~(size_t)255; };
  char* p = (char*)d_ws;
  float* deg = (float*)p;  p += al((size_t)N * 4);
  float* agg = (float*)p;  p += al((size_t)N * 128 * 4);
  float* h1  = (float*)p;  p += al((size_t)N * 128 * 4);
  float* h2  = (float*)p;  p += al((size_t)N * 128 * 4);
  unsigned short* wb1n = (unsigned short*)p; p += al(128 * 128 * 2);
  unsigned short* wb1r = (unsigned short*)p; p += al(128 * 128 * 2);
  unsigned short* wb2n = (unsigned short*)p; p += al(128 * 128 * 2);
  unsigned short* wb2r = (unsigned short*)p; p += al(128 * 128 * 2);
  unsigned short* wb3n = (unsigned short*)p; p += al(64 * 128 * 2);
  unsigned short* wb3r = (unsigned short*)p; p += al(64 * 128 * 2);

  const long long aggN = (long long)N * 128;
  const int zb  = (int)((aggN + 255) / 256);
  const int nb  = (N + 255) / 256;
  const int eb  = (E + 255) / 256;
  const int scTotal = E * 32;                      // (edge, 4-channel) items
  const int scb = (scTotal + 255) / 256;
  const int gb  = (N + 15) / 16;                   // one 16-row wave per block

  // degrees -> inverse degrees (once)
  zero_f32_kernel<<<nb, 256, 0, stream>>>(deg, (long long)N);
  degree_kernel<<<eb, 256, 0, stream>>>(dst, deg, E);
  invdeg_kernel<<<nb, 256, 0, stream>>>(deg, N);

  // bf16 weight conversions (once); pairwise packed
  cvt_bf16_kernel<<<(128 * 64 + 255) / 256, 256, 0, stream>>>(w1n, (unsigned int*)wb1n, 128 * 64);
  cvt_bf16_kernel<<<(128 * 64 + 255) / 256, 256, 0, stream>>>(w1r, (unsigned int*)wb1r, 128 * 64);
  cvt_bf16_kernel<<<(128 * 64 + 255) / 256, 256, 0, stream>>>(w2n, (unsigned int*)wb2n, 128 * 64);
  cvt_bf16_kernel<<<(128 * 64 + 255) / 256, 256, 0, stream>>>(w2r, (unsigned int*)wb2r, 128 * 64);
  cvt_bf16_kernel<<<(64 * 64 + 255) / 256, 256, 0, stream>>>(w3n, (unsigned int*)wb3n, 64 * 64);
  cvt_bf16_kernel<<<(64 * 64 + 255) / 256, 256, 0, stream>>>(w3r, (unsigned int*)wb3r, 64 * 64);

  // ---- layer 1: x -> h1 (relu)
  zero_f32_kernel<<<zb, 256, 0, stream>>>(agg, aggN);
  scatter_add_kernel<<<scb, 256, 0, stream>>>(x, src, dst, agg, scTotal);
  sage_wmma_kernel<128, true><<<gb, 32, 0, stream>>>(agg, x, deg, wb1n, wb1r, b1, h1, N);

  // ---- layer 2: h1 -> h2 (relu)
  zero_f32_kernel<<<zb, 256, 0, stream>>>(agg, aggN);
  scatter_add_kernel<<<scb, 256, 0, stream>>>(h1, src, dst, agg, scTotal);
  sage_wmma_kernel<128, true><<<gb, 32, 0, stream>>>(agg, h1, deg, wb2n, wb2r, b2, h2, N);

  // ---- layer 3: h2 -> out (no relu, COUT=64)
  zero_f32_kernel<<<zb, 256, 0, stream>>>(agg, aggN);
  scatter_add_kernel<<<scb, 256, 0, stream>>>(h2, src, dst, agg, scTotal);
  sage_wmma_kernel<64, false><<<gb, 32, 0, stream>>>(agg, h2, deg, wb3n, wb3r, b3,
                                                     (float*)d_out, N);
}
